// SelfAttention_68771016343750
// MI455X (gfx1250) — matrix operand order repaired and checked by reading, not compiled, not run on hardware
//
#include <hip/hip_runtime.h>

// ---------------------------------------------------------------------------
// Fused per-token self-attention for MI455X (gfx1250, wave32, WMMA).
//   qkv = x @ Wqkv + b ; per-token 16x16 head attention ; out = o @ Wout + b
// All GEMMs use v_wmma_f32_16x16x32_bf16. QKV intermediate stays in LDS.
// ---------------------------------------------------------------------------

#define D_MODEL 1024
#define QKV_N   3072
#define TOK_WG  32            // tokens per workgroup
#define STRX    1032          // LDS row stride (halves) for X / O tiles
#define STRQ    3080          // LDS row stride (halves) for QKV tiles

#define SMEM_X  (TOK_WG * STRX * 2)              // 66048 B
#define SMEM_Q  (TOK_WG * STRQ * 2)              // 197120 B
#define SMEM_A  (8 * 256 * 2)                    // 4096 B  (per-wave attn scratch)
#define SMEM_TOTAL (SMEM_X + SMEM_Q + SMEM_A)    // 267264 B  (<= 320KB WGP LDS)

typedef __attribute__((ext_vector_type(16))) __bf16 v16bf;
typedef __attribute__((ext_vector_type(8)))  float  v8f;

union BF16Tile { v16bf v; uint4 q[2]; };

static __device__ __forceinline__ unsigned int f2bf(float f) {
  union { float f; unsigned int u; } x; x.f = f;
  unsigned int r = x.u + 0x7FFFu + ((x.u >> 16) & 1u);   // round-to-nearest-even
  return r >> 16;                                        // bf16 bits in [15:0]
}

// ---------------------------------------------------------------------------
// Pre-swizzle an fp32 KxN weight matrix into bf16 WMMA B-operand tiles.
// Tile (kt,nt) of 32x16 is stored as 32 lanes x 32 bytes contiguous:
//   dword d of lane L holds  W[kt*32 + (L>=16)*8 + (d&3)*2 + (d>>2)*16 + p][nt*16 + L%16]
// so a wave loads one B tile as two contiguous b128 per lane.
// ---------------------------------------------------------------------------
__global__ __launch_bounds__(256)
void pack_wmma_b(const float* __restrict__ W, unsigned int* __restrict__ dst,
                 int K, int N) {
  int gid = blockIdx.x * 256 + threadIdx.x;
  int total = (K * N) >> 1;
  if (gid >= total) return;
  int d = gid & 7;
  int L = (gid >> 3) & 31;
  int t = gid >> 8;
  int ktiles = K >> 5;
  int kt = t % ktiles;
  int nt = t / ktiles;
  int n  = nt * 16 + (L & 15);
  int k0 = kt * 32 + ((L >> 4) << 3) + ((d & 3) << 1) + ((d >> 2) << 4);
  unsigned int lo = f2bf(W[(size_t)k0 * N + n]);
  unsigned int hi = f2bf(W[(size_t)(k0 + 1) * N + n]);
  dst[gid] = lo | (hi << 16);
}

// ---------------------------------------------------------------------------
// Fused attention: 1 workgroup = 32 tokens, 8 waves.
// ---------------------------------------------------------------------------
__global__ __launch_bounds__(256)
void attn_fused(const float* __restrict__ x,
                const unsigned int* __restrict__ WqkvP,
                const float* __restrict__ b_qkv,
                const unsigned int* __restrict__ WoutP,
                const float* __restrict__ b_out,
                float* __restrict__ out) {
  extern __shared__ char smem[];
  unsigned short* ldsX = (unsigned short*)smem;                   // 32 x 1024 (stride 1032) bf16; reused as O tile
  unsigned short* ldsQ = (unsigned short*)(smem + SMEM_X);        // 32 x 3072 (stride 3080) bf16 QKV
  unsigned short* ldsA = (unsigned short*)(smem + SMEM_X + SMEM_Q); // 8 x (16x16) attn scratch

  const int tid   = threadIdx.x;
  const int wave  = tid >> 5;
  const int lane  = tid & 31;
  const int lhalf = lane & 15;   // column index n / row m within half-wave
  const int hi16  = lane >> 4;   // which 16-lane half
  const int tokBase = blockIdx.x * TOK_WG;

  // ---- Phase 0: stage x tile (32 x 1024 fp32 -> bf16 in LDS) --------------
  {
    const float4* xrow = (const float4*)(x + (size_t)tokBase * D_MODEL);
    for (int i = tid; i < TOK_WG * (D_MODEL / 4); i += 256) {
      float4 v = xrow[i];
      int row = i >> 8;          // 256 float4 per row
      int c4  = i & 255;
      unsigned int* dst = (unsigned int*)(ldsX + row * STRX + c4 * 4);
      dst[0] = f2bf(v.x) | (f2bf(v.y) << 16);
      dst[1] = f2bf(v.z) | (f2bf(v.w) << 16);
    }
  }
  __syncthreads();

  // ---- Phase 1: QKV GEMM  (2 M-tiles x 192 N-tiles, K = 1024) -------------
  for (int tile = wave; tile < 2 * (QKV_N / 16); tile += 8) {
    int mt = tile / (QKV_N / 16);
    int nt = tile % (QKV_N / 16);
    v8f acc = {0.f, 0.f, 0.f, 0.f, 0.f, 0.f, 0.f, 0.f};
    unsigned rowoff = (unsigned)(mt * 16 + lhalf) * STRX;
    const uint4* bp = (const uint4*)WqkvP + (size_t)(nt * 32) * 64 + lane * 2;
    for (int kt = 0; kt < 32; ++kt) {
      BF16Tile A, B;
      const uint4* ap = (const uint4*)(ldsX + rowoff + kt * 32 + hi16 * 8);
      A.q[0] = ap[0];            // k = hi16*8 + 0..7
      A.q[1] = ap[2];            // k = 16 + hi16*8 + 0..7
      B.q[0] = bp[0];
      B.q[1] = bp[1];
      if (kt + 1 < 32) __builtin_prefetch((const void*)(bp + 64), 0, 0);
      bp += 64;
      acc = __builtin_amdgcn_wmma_f32_16x16x32_bf16(false, A.v, false, B.v,
                                                    (short)0, acc, false, false);
    }
    float bias = b_qkv[nt * 16 + lhalf];
    int col = nt * 16 + lhalf;
    for (int r = 0; r < 8; ++r) {
      int row = mt * 16 + r + hi16 * 8;
      ldsQ[row * STRQ + col] = (unsigned short)f2bf(acc[r] + bias);
    }
  }
  __syncthreads();

  // ---- Phase 2/3: per-token head attention (4 tokens per wave) ------------
  {
    const float scale = 0.125f;                 // 1/sqrt(64)
    unsigned short* aw = ldsA + wave * 256;     // 16x16 bf16 scratch per wave
    for (int t = wave * 4; t < wave * 4 + 4; ++t) {
      unsigned qbase = (unsigned)t * STRQ;
      // S = q @ k^T  (16 heads x 16 heads, K = 64 -> two WMMAs)
      v8f c = {0.f, 0.f, 0.f, 0.f, 0.f, 0.f, 0.f, 0.f};
      for (int k0 = 0; k0 < 64; k0 += 32) {
        BF16Tile A, B;
        const uint4* ap = (const uint4*)(ldsQ + qbase + lhalf * 64 + k0 + hi16 * 8);
        A.q[0] = ap[0];
        A.q[1] = ap[2];
        const uint4* kp = (const uint4*)(ldsQ + qbase + D_MODEL + lhalf * 64 + k0 + hi16 * 8);
        B.q[0] = kp[0];
        B.q[1] = kp[2];
        c = __builtin_amdgcn_wmma_f32_16x16x32_bf16(false, A.v, false, B.v,
                                                    (short)0, c, false, false);
      }
      // softmax over the 16 columns: rows live across the 16-lane halves,
      // xor masks 1/2/4/8 never cross the half boundary.
      for (int r = 0; r < 8; ++r) {
        float p = c[r] * scale;
        float mx = p;
        for (int m = 8; m >= 1; m >>= 1) mx = fmaxf(mx, __shfl_xor(mx, m, 32));
        float e = __expf(p - mx);
        float s = e;
        for (int m = 8; m >= 1; m >>= 1) s += __shfl_xor(s, m, 32);
        float a = e / s;
        aw[(r + hi16 * 8) * 16 + lhalf] = (unsigned short)f2bf(a);
      }
      asm volatile("s_wait_dscnt 0" ::: "memory");   // same-wave LDS bounce
      // O = attn @ v  (attn zero-padded to K=32; only k<16 is live)
      BF16Tile Aa;
      const uint4* aap = (const uint4*)(aw + lhalf * 16);
      Aa.q[0] = hi16 ? make_uint4(0u, 0u, 0u, 0u) : aap[0];
      Aa.q[1] = hi16 ? make_uint4(0u, 0u, 0u, 0u) : aap[1];
      for (int nt = 0; nt < 4; ++nt) {
        BF16Tile Bv;
        unsigned vbase = qbase + 2 * D_MODEL + nt * 16 + lhalf;
        unsigned int bw[8];
        for (int r = 0; r < 8; ++r) {
          unsigned int lo  = ldsQ[vbase + (2 * r) * 64];
          unsigned int hiv = ldsQ[vbase + (2 * r + 1) * 64];
          bw[r] = lo | (hiv << 16);
        }
        Bv.q[0] = make_uint4(bw[0], bw[1], bw[2], bw[3]);
        Bv.q[1] = make_uint4(bw[4], bw[5], bw[6], bw[7]);
        v8f o = {0.f, 0.f, 0.f, 0.f, 0.f, 0.f, 0.f, 0.f};
        o = __builtin_amdgcn_wmma_f32_16x16x32_bf16(false, Aa.v, false, Bv.v,
                                                    (short)0, o, false, false);
        for (int r = 0; r < 8; ++r) {
          int i = r + hi16 * 8;                       // head index
          ldsX[t * STRX + i * 64 + nt * 16 + lhalf] = (unsigned short)f2bf(o[r]);
        }
      }
    }
  }
  __syncthreads();

  // ---- Phase 4: out projection (2 M-tiles x 64 N-tiles, K = 1024) ---------
  for (int tile = wave; tile < 2 * (D_MODEL / 16); tile += 8) {
    int mt = tile / (D_MODEL / 16);
    int nt = tile % (D_MODEL / 16);
    v8f acc = {0.f, 0.f, 0.f, 0.f, 0.f, 0.f, 0.f, 0.f};
    unsigned rowoff = (unsigned)(mt * 16 + lhalf) * STRX;
    const uint4* bp = (const uint4*)WoutP + (size_t)(nt * 32) * 64 + lane * 2;
    for (int kt = 0; kt < 32; ++kt) {
      BF16Tile A, B;
      const uint4* ap = (const uint4*)(ldsX + rowoff + kt * 32 + hi16 * 8);
      A.q[0] = ap[0];
      A.q[1] = ap[2];
      B.q[0] = bp[0];
      B.q[1] = bp[1];
      if (kt + 1 < 32) __builtin_prefetch((const void*)(bp + 64), 0, 0);
      bp += 64;
      acc = __builtin_amdgcn_wmma_f32_16x16x32_bf16(false, A.v, false, B.v,
                                                    (short)0, acc, false, false);
    }
    float bias = b_out[nt * 16 + lhalf];
    int col = nt * 16 + lhalf;
    for (int r = 0; r < 8; ++r) {
      int row = tokBase + mt * 16 + r + hi16 * 8;
      out[(size_t)row * D_MODEL + col] = acc[r] + bias;
    }
  }
}

// ---------------------------------------------------------------------------
extern "C" void kernel_launch(void* const* d_in, const int* in_sizes, int n_in,
                              void* d_out, int out_size, void* d_ws, size_t ws_size,
                              hipStream_t stream) {
  const float* x     = (const float*)d_in[0];
  const float* W_qkv = (const float*)d_in[1];
  const float* b_qkv = (const float*)d_in[2];
  const float* W_out = (const float*)d_in[3];
  const float* b_out = (const float*)d_in[4];
  float* out = (float*)d_out;

  // workspace: pre-swizzled bf16 weights (6 MB + 2 MB)
  unsigned int* WqkvP = (unsigned int*)d_ws;
  unsigned int* WoutP = WqkvP + (D_MODEL * QKV_N) / 2;

  {
    int total = (D_MODEL * QKV_N) / 2;
    pack_wmma_b<<<(total + 255) / 256, 256, 0, stream>>>(W_qkv, WqkvP, D_MODEL, QKV_N);
  }
  {
    int total = (D_MODEL * D_MODEL) / 2;
    pack_wmma_b<<<(total + 255) / 256, 256, 0, stream>>>(W_out, WoutP, D_MODEL, D_MODEL);
  }

  const int nTok = 4 * 4096;       // B * S
  attn_fused<<<nTok / TOK_WG, 256, SMEM_TOTAL, stream>>>(
      x, WqkvP, b_qkv, WoutP, b_out, out);
}
